// EGNNvelBaseline_69818988364013
// MI455X (gfx1250) — matrix operand order repaired and checked by reading, not compiled, not run on hardware
//
#include <hip/hip_runtime.h>
#include <hip/hip_bf16.h>
#include <cmath>

#define VNODES 16000
#define NN     125
#define NEDGE  124
#define HID    64
#define KPAD   96      // padded K for GEMM1: h_col(64)+radial(1)+ea(2)=67 -> 96
#define EPG    15500   // edges per graph = N*(N-1)
#define LWSTRIDE (64*KPAD + 2*64*64)   // f16 weights per layer: W0T + W1T + Wc0T

typedef __attribute__((ext_vector_type(16))) _Float16 v16h;
typedef __attribute__((ext_vector_type(8)))  _Float16 h8;
typedef __attribute__((ext_vector_type(4)))  _Float16 h4;
typedef __attribute__((ext_vector_type(2)))  _Float16 h2;
typedef __attribute__((ext_vector_type(8)))  float    v8f;

// Fast SiLU: x * rcp(1 + exp2(-x*log2(e))) -> v_exp_f32 + v_rcp_f32 (hardware rate)
__device__ __forceinline__ float silu_f(float x) {
  float e = __builtin_amdgcn_exp2f(x * -1.44269504088896340736f);
  return x * __builtin_amdgcn_rcpf(1.0f + e);
}

// A-fragment (16x32 f16): lane l holds row m=l&15; halves 0..7 = K kb..kb+7,
// halves 8..15 = K kb+16..kb+23, with kb = (l<16)?0:8 inside a 32-wide K tile.
__device__ __forceinline__ v16h ld_fragA(const _Float16* p) {
  h8 lo = *(const h8*)p;
  h8 hi = *(const h8*)(p + 16);
  return __builtin_shufflevector(lo, hi, 0,1,2,3,4,5,6,7,8,9,10,11,12,13,14,15);
}
// B-fragment (32x16 f16): lane l holds col n=l&15; halves 0..7 = K kb..kb+7,
// halves 8..15 = K kb+16..kb+23, kb = (l<16)?0:8 (weights stored transposed [N][Kpad]).
__device__ __forceinline__ v16h ld_fragB(const _Float16* p) {
  h8 lo = *(const h8*)p;
  h8 hi = *(const h8*)(p + 16);
  return __builtin_shufflevector(lo, hi, 0,1,2,3,4,5,6,7,8,9,10,11,12,13,14,15);
}
#define WMMA_F16(a,b,c) __builtin_amdgcn_wmma_f32_16x16x32_f16(false,(a),false,(b),(short)0,(c),false,false)

// ---------------- weight prep: f32 [K][N] -> f16 transposed [N][Kpad] ----------------
__global__ void egnn_prep(const float* __restrict__ em0W, const float* __restrict__ em1W,
                          const float* __restrict__ cm0W, _Float16* __restrict__ dst) {
  int i = blockIdx.x * blockDim.x + threadIdx.x;
  if (i < 64*KPAD) {                     // W0T: rows 64..130 of edge_mlp[0].W, zero-padded
    int n = i / KPAD, kk = i % KPAD;
    dst[i] = (kk < 67) ? (_Float16)em0W[(64 + kk)*HID + n] : (_Float16)0.0f;
  } else if (i < 64*KPAD + 64*64) {      // W1T
    int t = i - 64*KPAD; int n = t / 64, k = t % 64;
    dst[i] = (_Float16)em1W[k*HID + n];
  } else if (i < LWSTRIDE) {             // Wc0T
    int t = i - 64*KPAD - 64*64; int n = t / 64, k = t % 64;
    dst[i] = (_Float16)cm0W[k*HID + n];
  }
}

// ---------------- embedding: h = nodes @ We + be ----------------
__global__ void egnn_embed(const float* __restrict__ nodes, const float* __restrict__ W,
                           const float* __restrict__ bb, float* __restrict__ h) {
  int i = blockIdx.x * blockDim.x + threadIdx.x;
  if (i >= VNODES * HID) return;
  int r = i >> 6, j = i & 63;
  float a = bb[j];
  #pragma unroll
  for (int k = 0; k < 8; ++k) a += nodes[r*8 + k] * W[k*HID + j];
  h[i] = a;
}

// ---------------- projection: out = h @ Wp + bp ----------------
__global__ void egnn_proj(const float* __restrict__ h, const float* __restrict__ W,
                          const float* __restrict__ bb, float* __restrict__ out) {
  int i = blockIdx.x * blockDim.x + threadIdx.x;
  if (i >= VNODES * 8) return;
  int r = i >> 3, p = i & 7;
  float a = bb[p];
  #pragma unroll 8
  for (int k = 0; k < HID; ++k) a += h[r*HID + k] * W[k*8 + p];
  out[i] = a;
}

// ---------------- fused EGNN layer: one block (128 thr = 4 waves) per node ----------------
__global__ __launch_bounds__(128) void egnn_layer(
    const float* __restrict__ hIn, const float* __restrict__ xIn, const float* __restrict__ vIn,
    const float* __restrict__ edge_attr,
    const _Float16* __restrict__ W0T, const _Float16* __restrict__ W1T, const _Float16* __restrict__ Wc0T,
    const float* __restrict__ em0W, const float* __restrict__ em0b, const float* __restrict__ em1b,
    const float* __restrict__ cm0b, const float* __restrict__ cm1W,
    const float* __restrict__ cmv0W, const float* __restrict__ cmv0b,
    const float* __restrict__ cmv1W, const float* __restrict__ cmv1b,
    const float* __restrict__ nm0W, const float* __restrict__ nm0b,
    const float* __restrict__ nm1W, const float* __restrict__ nm1b,
    float* __restrict__ hOut, float* __restrict__ xOut, float* __restrict__ vOut)
{
  __shared__ float s_hrow[HID];
  __shared__ float s_vrow[HID];       // h[row] @ W0[:64] + b0  (rank-1 part of GEMM1)
  __shared__ float s_wc1[HID];
  __shared__ float s_maggP[4][HID];
  __shared__ float s_taccP[4][4];
  __shared__ float s_magg[HID];
  __shared__ float s_agg[3];
  __shared__ float s_hidn[HID];
  __shared__ float s_hidv[HID];
  __shared__ float s_velscale;
  __shared__ alignas(16) _Float16 s_X [4][16*KPAD];  // per-wave GEMM1 input
  __shared__ alignas(16) _Float16 s_H1[4][16*HID];
  __shared__ alignas(16) _Float16 s_M [4][16*HID];
  __shared__ alignas(16) _Float16 s_C1[4][16*HID];

  const int r = blockIdx.x;
  const int b = r / NN, rr = r % NN;
  const int tid = threadIdx.x;
  const int wave = tid >> 5, lane = tid & 31;

  if (tid < HID) { s_hrow[tid] = hIn[r*HID + tid]; s_wc1[tid] = cm1W[tid]; }
  __syncthreads();
  if (tid < HID) {
    float acc = em0b[tid];
    #pragma unroll 8
    for (int k = 0; k < HID; ++k) acc += s_hrow[k] * em0W[k*HID + tid];
    s_vrow[tid] = acc;
  }

  const float xr0 = xIn[r*3+0], xr1 = xIn[r*3+1], xr2 = xIn[r*3+2];

  _Float16* Xw  = &s_X [wave][0];
  _Float16* H1w = &s_H1[wave][0];
  _Float16* Mw  = &s_M [wave][0];
  _Float16* C1w = &s_C1[wave][0];

  const int eSlot = lane & 15;          // edge slot within tile
  const int half  = lane >> 4;          // which 32-feature half of h[col]
  const int nIdx  = lane & 15;
  const int mbase = (lane < 16) ? 0 : 8;
  const int kbA   = (lane < 16) ? 0 : 8;
  const int kbB   = (lane < 16) ? 0 : 8;

  float tacc0 = 0.f, tacc1 = 0.f, tacc2 = 0.f;  // trans accumulator (lanes 0-15)
  float macc0 = 0.f, macc1 = 0.f;               // m_agg, 2 columns per lane

  for (int it = 0; it < 2; ++it) {
    const int t  = wave*2 + it;
    const int k0 = t*16;
    const int valid = (NEDGE - k0 < 16) ? (NEDGE - k0) : 16;
    float cd0 = 0.f, cd1 = 0.f, cd2 = 0.f;

    // ---- stage X = [h[col] | radial | ea | 0-pad] as f16, 16 edges x KPAD ----
    {
      const int k = k0 + eSlot;
      if (eSlot < valid) {
        const int colLocal = (k < rr) ? k : k + 1;
        const int col = b*NN + colLocal;
        const float* hp = hIn + col*HID + half*32;
        #pragma unroll
        for (int f = 0; f < 32; f += 4) {
          float4 hv = *(const float4*)(hp + f);
          h4 o = { (_Float16)hv.x, (_Float16)hv.y, (_Float16)hv.z, (_Float16)hv.w };
          *(h4*)(Xw + eSlot*KPAD + half*32 + f) = o;
        }
        if (half == 0) {
          cd0 = xr0 - xIn[col*3+0];
          cd1 = xr1 - xIn[col*3+1];
          cd2 = xr2 - xIn[col*3+2];
          const float radial = cd0*cd0 + cd1*cd1 + cd2*cd2;
          const int eidx = b*EPG + rr*NEDGE + k;
          Xw[eSlot*KPAD + 64] = (_Float16)radial;
          Xw[eSlot*KPAD + 65] = (_Float16)edge_attr[2*eidx + 0];
          Xw[eSlot*KPAD + 66] = (_Float16)edge_attr[2*eidx + 1];
          for (int f = 67; f < KPAD; ++f) Xw[eSlot*KPAD + f] = (_Float16)0.0f;
        }
      } else {
        h4 z = {};
        #pragma unroll
        for (int f = 0; f < 32; f += 4) *(h4*)(Xw + eSlot*KPAD + half*32 + f) = z;
        if (half == 0)
          for (int f = 64; f < KPAD; ++f) Xw[eSlot*KPAD + f] = (_Float16)0.0f;
      }
    }
    __syncthreads();

    // ---- GEMM1: X[16x96] @ W0T -> +vrow, SiLU -> H1 ----
    #pragma unroll
    for (int nt = 0; nt < 4; ++nt) {
      v8f c = {};
      #pragma unroll
      for (int kt = 0; kt < 3; ++kt) {
        v16h a  = ld_fragA(Xw + nIdx*KPAD + kt*32 + kbA);
        v16h bf = ld_fragB(W0T + (nt*16 + nIdx)*KPAD + kt*32 + kbB);
        c = WMMA_F16(a, bf, c);
      }
      const int n = nt*16 + nIdx;
      const float add = s_vrow[n];
      #pragma unroll
      for (int j = 0; j < 8; ++j)
        H1w[(mbase + j)*HID + n] = (_Float16)silu_f(c[j] + add);
    }
    __syncthreads();

    // ---- GEMM2: H1 @ W1T -> +b1, SiLU -> m ----
    #pragma unroll
    for (int nt = 0; nt < 4; ++nt) {
      v8f c = {};
      #pragma unroll
      for (int kt = 0; kt < 2; ++kt) {
        v16h a  = ld_fragA(H1w + nIdx*HID + kt*32 + kbA);
        v16h bf = ld_fragB(W1T + (nt*16 + nIdx)*HID + kt*32 + kbB);
        c = WMMA_F16(a, bf, c);
      }
      const int n = nt*16 + nIdx;
      const float add = em1b[n];
      #pragma unroll
      for (int j = 0; j < 8; ++j)
        Mw[(mbase + j)*HID + n] = (_Float16)silu_f(c[j] + add);
    }
    __syncthreads();

    // ---- GEMM3: m @ Wc0T -> +bc0, SiLU -> C1 ----
    #pragma unroll
    for (int nt = 0; nt < 4; ++nt) {
      v8f c = {};
      #pragma unroll
      for (int kt = 0; kt < 2; ++kt) {
        v16h a  = ld_fragA(Mw + nIdx*HID + kt*32 + kbA);
        v16h bf = ld_fragB(Wc0T + (nt*16 + nIdx)*HID + kt*32 + kbB);
        c = WMMA_F16(a, bf, c);
      }
      const int n = nt*16 + nIdx;
      const float add = cm0b[n];
      #pragma unroll
      for (int j = 0; j < 8; ++j)
        C1w[(mbase + j)*HID + n] = (_Float16)silu_f(c[j] + add);
    }
    __syncthreads();

    // ---- per-edge scalar w = C1 . wc1 (no bias), trans accumulation ----
    if (lane < 16 && eSlot < valid) {
      float w = 0.f;
      #pragma unroll 8
      for (int n = 0; n < HID; ++n) w += (float)C1w[eSlot*HID + n] * s_wc1[n];
      tacc0 += cd0 * w; tacc1 += cd1 * w; tacc2 += cd2 * w;
    }
    // ---- m_agg accumulation: each lane sums 2 columns over valid edges ----
    for (int e = 0; e < valid; ++e) {
      h2 mv = *(const h2*)(Mw + e*HID + 2*lane);
      macc0 += (float)mv.x; macc1 += (float)mv.y;
    }
  }

  // wave-level reduce of trans (lanes 16-31 hold zeros)
  #pragma unroll
  for (int off = 16; off >= 1; off >>= 1) {
    tacc0 += __shfl_xor(tacc0, off, 32);
    tacc1 += __shfl_xor(tacc1, off, 32);
    tacc2 += __shfl_xor(tacc2, off, 32);
  }
  if (lane == 0) { s_taccP[wave][0] = tacc0; s_taccP[wave][1] = tacc1; s_taccP[wave][2] = tacc2; }
  s_maggP[wave][2*lane + 0] = macc0;
  s_maggP[wave][2*lane + 1] = macc1;
  __syncthreads();

  if (tid < HID)
    s_magg[tid] = s_maggP[0][tid] + s_maggP[1][tid] + s_maggP[2][tid] + s_maggP[3][tid];
  if (tid < 3)
    s_agg[tid] = (s_taccP[0][tid] + s_taccP[1][tid] + s_taccP[2][tid] + s_taccP[3][tid])
                 * (1.0f / (float)NEDGE);   // segment mean (cnt == 124 for all nodes)
  __syncthreads();

  // ---- node model hiddens + velocity-MLP hidden (f32 VALU) ----
  if (tid < HID) {
    float av = cmv0b[tid];
    float an = nm0b[tid];
    #pragma unroll 4
    for (int k = 0; k < HID; ++k) {
      av += s_hrow[k] * cmv0W[k*HID + tid];
      an += s_hrow[k] * nm0W[k*HID + tid];
    }
    #pragma unroll 4
    for (int k = 0; k < HID; ++k) an += s_magg[k] * nm0W[(HID + k)*HID + tid];
    s_hidv[tid] = silu_f(av);
    s_hidn[tid] = silu_f(an);
  }
  __syncthreads();
  if (tid == 0) {
    float vs = cmv1b[0];
    for (int j = 0; j < HID; ++j) vs += s_hidv[j] * cmv1W[j];
    s_velscale = vs;
  }
  __syncthreads();
  if (tid < HID) {
    float o = nm1b[tid];
    #pragma unroll 4
    for (int j = 0; j < HID; ++j) o += s_hidn[j] * nm1W[j*HID + tid];
    hOut[r*HID + tid] = s_hrow[tid] + o;          // recurrent residual
  }
  if (tid < 3) {
    const float vn = s_velscale * vIn[r*3 + tid] + s_agg[tid];
    vOut[r*3 + tid] = vn;
    xOut[r*3 + tid] = xIn[r*3 + tid] + vn;
  }
}

extern "C" void kernel_launch(void* const* d_in, const int* in_sizes, int n_in,
                              void* d_out, int out_size, void* d_ws, size_t ws_size,
                              hipStream_t stream) {
  // jax pytree leaf order (sorted dict keys): embedding{W,b}, layers[0..3]{
  //   coord_mlp[0]{W,b},coord_mlp[1]{W}, coord_mlp_vel[0]{W,b},coord_mlp_vel[1]{W,b},
  //   edge_mlp[0]{W,b},edge_mlp[1]{W,b}, node_mlp[0]{W,b},node_mlp[1]{W,b} }, projection{W,b}
  const float* nodes     = (const float*)d_in[0];
  const float* loc       = (const float*)d_in[1];
  const float* vel       = (const float*)d_in[2];
  const float* edge_attr = (const float*)d_in[4];
  const float* embW      = (const float*)d_in[5];
  const float* embB      = (const float*)d_in[6];
  const float* projW     = (const float*)d_in[67];
  const float* projB     = (const float*)d_in[68];

  const float *cm0W[4], *cm0b[4], *cm1W[4], *cmv0W[4], *cmv0b[4], *cmv1W[4], *cmv1b[4];
  const float *em0W[4], *em0b[4], *em1W[4], *em1b[4], *nm0W[4], *nm0b[4], *nm1W[4], *nm1b[4];
  for (int l = 0; l < 4; ++l) {
    int p = 7 + l*15;
    cm0W[l]  = (const float*)d_in[p+0];  cm0b[l]  = (const float*)d_in[p+1];
    cm1W[l]  = (const float*)d_in[p+2];
    cmv0W[l] = (const float*)d_in[p+3];  cmv0b[l] = (const float*)d_in[p+4];
    cmv1W[l] = (const float*)d_in[p+5];  cmv1b[l] = (const float*)d_in[p+6];
    em0W[l]  = (const float*)d_in[p+7];  em0b[l]  = (const float*)d_in[p+8];
    em1W[l]  = (const float*)d_in[p+9];  em1b[l]  = (const float*)d_in[p+10];
    nm0W[l]  = (const float*)d_in[p+11]; nm0b[l]  = (const float*)d_in[p+12];
    nm1W[l]  = (const float*)d_in[p+13]; nm1b[l]  = (const float*)d_in[p+14];
  }

  // workspace layout (all 16B aligned)
  float* hA = (float*)d_ws;
  float* hB = hA + (size_t)VNODES*HID;
  float* xA = hB + (size_t)VNODES*HID;
  float* xB = xA + (size_t)VNODES*3;
  float* vA = xB + (size_t)VNODES*3;
  float* vB = vA + (size_t)VNODES*3;
  _Float16* wbuf = (_Float16*)(vB + (size_t)VNODES*3);

  for (int l = 0; l < 4; ++l)
    egnn_prep<<<(LWSTRIDE + 255)/256, 256, 0, stream>>>(em0W[l], em1W[l], cm0W[l],
                                                        wbuf + (size_t)l*LWSTRIDE);
  egnn_embed<<<(VNODES*HID + 255)/256, 256, 0, stream>>>(nodes, embW, embB, hA);

  float* out_h = (float*)d_out;
  float* out_x = out_h + (size_t)VNODES*8;
  float* out_v = out_x + (size_t)VNODES*3;

  const float* hin = hA; const float* xin = loc; const float* vin = vel;
  float* hbufs[2] = {hB, hA};
  float* xbufs[2] = {xA, xB};
  float* vbufs[2] = {vA, vB};
  for (int l = 0; l < 4; ++l) {
    _Float16* wl = wbuf + (size_t)l*LWSTRIDE;
    float* hout = hbufs[l & 1];
    float* xout = (l == 3) ? out_x : xbufs[l & 1];
    float* vout = (l == 3) ? out_v : vbufs[l & 1];
    egnn_layer<<<VNODES, 128, 0, stream>>>(hin, xin, vin, edge_attr,
        wl, wl + 64*KPAD, wl + 64*KPAD + 64*64,
        em0W[l], em0b[l], em1b[l], cm0b[l], cm1W[l],
        cmv0W[l], cmv0b[l], cmv1W[l], cmv1b[l],
        nm0W[l], nm0b[l], nm1W[l], nm1b[l],
        hout, xout, vout);
    hin = hout; xin = xout; vin = vout;
  }
  egnn_proj<<<(VNODES*8 + 255)/256, 256, 0, stream>>>(hin, projW, projB, out_h);
}
